// GlobalPushPlus_20684562497828
// MI455X (gfx1250) — matrix operand order repaired and checked by reading, not compiled, not run on hardware
//
#include <hip/hip_runtime.h>

typedef __attribute__((ext_vector_type(16))) __bf16    v16bf;
typedef __attribute__((ext_vector_type(8)))  float     v8f;
typedef __attribute__((ext_vector_type(8)))  unsigned  v8u;

#define MARGIN_F 0.15001125f   // 0.15 + (1/40000)*(0.6-0.15)
#define B_ROWS 8192
#define C_COLS 20000
#define C_PAD  20032           // padded to BT multiple: async tile loads never OOB
#define DDIM   128
#define BT     64              // block tile (M and N)
#define KSTAGE 64              // K staged through LDS
#define LDB    72              // LDS row stride in bf16 (144 B: 16B-aligned, bank-skewed)

__device__ __forceinline__ float waveSum(float v) {
#pragma unroll
  for (int off = 16; off; off >>= 1) v += __shfl_xor(v, off, 32);
  return v;
}
__device__ __forceinline__ unsigned waveSumU(unsigned v) {
#pragma unroll
  for (int off = 16; off; off >>= 1) v += __shfl_xor(v, off, 32);
  return v;
}

// Split 4 floats into packed bf16 hi (2 dwords) and bf16 lo (2 dwords).
// dword d = { bf16(k=2d) in [15:0], bf16(k=2d+1) in [31:16] }
__device__ __forceinline__ void cvt_pack(const float4& v, uint2& hi, uint2& lo) {
  unsigned u0 = __float_as_uint(v.x), u1 = __float_as_uint(v.y);
  unsigned u2 = __float_as_uint(v.z), u3 = __float_as_uint(v.w);
  unsigned h0 = u0 & 0xFFFF0000u, h1 = u1 & 0xFFFF0000u;
  unsigned h2 = u2 & 0xFFFF0000u, h3 = u3 & 0xFFFF0000u;
  float l0 = v.x - __uint_as_float(h0);
  float l1 = v.y - __uint_as_float(h1);
  float l2 = v.z - __uint_as_float(h2);
  float l3 = v.w - __uint_as_float(h3);
  hi.x = (u0 >> 16) | h1;
  hi.y = (u2 >> 16) | h3;
  lo.x = (__float_as_uint(l0) >> 16) | (__float_as_uint(l1) & 0xFFFF0000u);
  lo.y = (__float_as_uint(l2) >> 16) | (__float_as_uint(l3) & 0xFFFF0000u);
}

__device__ __forceinline__ v16bf frag_from(const unsigned short* p, int delta) {
  uint4 q0 = *(const uint4*)(p);
  uint4 q1 = *(const uint4*)(p + delta);   // delta in bf16 elements
  v8u r = {q0.x, q0.y, q0.z, q0.w, q1.x, q1.y, q1.z, q1.w};
  return __builtin_bit_cast(v16bf, r);
}

// CDNA5 async global->LDS copy (ASYNCcnt), bypasses VGPRs.
// LDS address = low 32 bits of the generic pointer (ISA 10.2).
__device__ __forceinline__ void async_b128(void* lds_ptr, const void* gptr) {
#if defined(__HIP_DEVICE_COMPILE__)
  asm volatile("global_load_async_to_lds_b128 %0, %1, off"
               :: "v"((unsigned)(uintptr_t)lds_ptr),
                  "v"((unsigned long long)(uintptr_t)gptr)
               : "memory");
#endif
}
__device__ __forceinline__ void wait_async0() {
#if defined(__HIP_DEVICE_COMPILE__)
  asm volatile("s_wait_asynccnt 0x0" ::: "memory");
#endif
}

#if defined(__HIP_DEVICE_COMPILE__)
#define WMMA_BF16(A, Bm, Cacc) \
  __builtin_amdgcn_wmma_f32_16x16x32_bf16(false, (A), false, (Bm), (short)0, (Cacc), false, false)
#else
#define WMMA_BF16(A, Bm, Cacc) (Cacc)
#endif

__global__ void init_accum(double* sum, unsigned* cnt) {
  *sum = 0.0;
  *cnt = 0u;
}

// One block (128 threads) per feature row: pos_dot[b] + normalized bf16 hi/lo row.
__global__ void prep_kernel(const float* __restrict__ feat,
                            const float* __restrict__ centers,
                            const int* __restrict__ labels,
                            float* __restrict__ pos,
                            unsigned short* __restrict__ Fhi,
                            unsigned short* __restrict__ Flo) {
  __shared__ float sred[4];
  const int b = blockIdx.x;
  const int d = threadIdx.x;                 // 128 threads == DDIM
  const int wid = d >> 5, lane = d & 31;

  float x = feat[b * DDIM + d];
  float p = waveSum(x * x);
  if (lane == 0) sred[wid] = p;
  __syncthreads();
  float ss = sred[0] + sred[1] + sred[2] + sred[3];
  __syncthreads();

  float iv = 1.0f / fmaxf(sqrtf(ss), 1e-12f);
  float xn = x * iv;
  // bf16 hi/lo split of normalized feature
  unsigned u = __float_as_uint(xn);
  unsigned h = u & 0xFFFF0000u;
  float fl = xn - __uint_as_float(h);
  Fhi[b * DDIM + d] = (unsigned short)(u >> 16);
  Flo[b * DDIM + d] = (unsigned short)(__float_as_uint(fl) >> 16);

  int lab = labels[b];
  float c = centers[lab * DDIM + d];
  float q = waveSum(xn * c);
  if (lane == 0) sred[wid] = q;
  __syncthreads();
  if (d == 0) pos[b] = sred[0] + sred[1] + sred[2] + sred[3];
}

// Convert centers -> packed bf16 hi/lo, zero-padded to C_PAD rows.
__global__ void cvt_centers_kernel(const float* __restrict__ centers,
                                   unsigned short* __restrict__ Chi,
                                   unsigned short* __restrict__ Clo) {
  int idx = blockIdx.x * 256 + threadIdx.x;   // float4 index over padded array
  int c  = idx >> 5;                          // 32 float4 per row
  int d4 = (idx & 31) << 2;
  float4 v = make_float4(0.f, 0.f, 0.f, 0.f);
  if (c < C_COLS) v = *(const float4*)&centers[c * DDIM + d4];
  uint2 h, l;
  cvt_pack(v, h, l);
  *(uint2*)&Chi[c * DDIM + d4] = h;
  *(uint2*)&Clo[c * DDIM + d4] = l;
}

__global__ void __launch_bounds__(256)
push_loss_kernel(const unsigned short* __restrict__ Fhi,
                 const unsigned short* __restrict__ Flo,
                 const unsigned short* __restrict__ Chi,
                 const unsigned short* __restrict__ Clo,
                 const int* __restrict__ labels,
                 const float* __restrict__ pos,
                 double* __restrict__ gsum,
                 unsigned* __restrict__ gcnt) {
  __shared__ unsigned short Ah[BT * LDB];
  __shared__ unsigned short Al[BT * LDB];
  __shared__ unsigned short Bh[BT * LDB];
  __shared__ unsigned short Bl[BT * LDB];
  __shared__ float    s_sum[8];
  __shared__ unsigned s_cnt[8];

  const int tid  = threadIdx.x;
  const int w    = tid >> 5;      // wave 0..7
  const int lane = tid & 31;
  const int lm   = lane & 15;
  const int lh   = lane >> 4;     // lane half: 0/1
  const int wm   = w & 3;         // which 16-row M tile
  const int wn   = w >> 2;        // which 32-col N group

  const int mbase = blockIdx.y * BT;
  const int cbase = blockIdx.x * BT;

  v8f acc0 = {};
  v8f acc1 = {};

  for (int s = 0; s < DDIM / KSTAGE; ++s) {
    const int ks = s * KSTAGE;
    // Async stage: 64 rows x 64 bf16 per array = 512 b128 copies, 2/thread/array.
#pragma unroll
    for (int i = 0; i < 2; ++i) {
      int f   = tid + i * 256;     // b128 index 0..511
      int row = f >> 3;            // 8 b128 per 64-bf16 row
      int c8  = (f & 7) << 3;      // bf16 offset within stage
      int lidx = row * LDB + c8;
      int ga = (mbase + row) * DDIM + ks + c8;
      int gb = (cbase + row) * DDIM + ks + c8;   // cbase+row < C_PAD always
      async_b128(&Ah[lidx], Fhi + ga);
      async_b128(&Al[lidx], Flo + ga);
      async_b128(&Bh[lidx], Chi + gb);
      async_b128(&Bl[lidx], Clo + gb);
    }
    wait_async0();
    __syncthreads();

#pragma unroll
    for (int kc = 0; kc < KSTAGE; kc += 32) {
      // A fragment 16x32 bf16: lane<16 -> M=lm, K = kc+{0..7,16..23};
      // lane>=16 -> M=lm, K = kc+{8..15,24..31}. Two b128 LDS loads.
      const int arow = (wm * 16 + lm) * LDB + kc + lh * 8;
      v16bf a_hi = frag_from(&Ah[arow], 16);
      v16bf a_lo = frag_from(&Al[arow], 16);
#pragma unroll
      for (int t = 0; t < 2; ++t) {
        // B fragment 32x16 bf16: lane -> column N=lm, K = kc + lh*8 + {0..7,16..23}
        const int brow = (wn * 32 + t * 16 + lm) * LDB + kc + lh * 8;
        v16bf b_hi = frag_from(&Bh[brow], 16);
        v16bf b_lo = frag_from(&Bl[brow], 16);
        if (t == 0) {
          acc0 = WMMA_BF16(a_hi, b_hi, acc0);
          acc0 = WMMA_BF16(a_hi, b_lo, acc0);
          acc0 = WMMA_BF16(a_lo, b_hi, acc0);
        } else {
          acc1 = WMMA_BF16(a_hi, b_hi, acc1);
          acc1 = WMMA_BF16(a_hi, b_lo, acc1);
          acc1 = WMMA_BF16(a_lo, b_hi, acc1);
        }
      }
    }
    __syncthreads();
  }

  // Fused epilogue: masked hinge sum + count (never materialize [B,C]).
  // C/D layout: VGPR i, lanes 0-15 -> M=i; lanes 16-31 -> M=i+8; N = lane%16.
  float lsum = 0.f;
  unsigned lcnt = 0;
#pragma unroll
  for (int i = 0; i < 8; ++i) {
    int m = mbase + wm * 16 + lh * 8 + i;
    float pv = pos[m];
    int lab = labels[m];
#pragma unroll
    for (int t = 0; t < 2; ++t) {
      int n = cbase + wn * 32 + t * 16 + lm;
      float dotv = (t == 0) ? acc0[i] : acc1[i];
      float loss = MARGIN_F + dotv - pv;
      if (n < C_COLS && lab != n && loss > 0.f) {
        lsum += loss;
        lcnt += 1u;
      }
    }
  }
  lsum = waveSum(lsum);
  lcnt = waveSumU(lcnt);
  if (lane == 0) { s_sum[w] = lsum; s_cnt[w] = lcnt; }
  __syncthreads();
  if (tid == 0) {
    double tot = 0.0;
    unsigned c = 0;
#pragma unroll
    for (int i = 0; i < 8; ++i) { tot += (double)s_sum[i]; c += s_cnt[i]; }
    atomicAdd(gsum, tot);
    atomicAdd(gcnt, c);
  }
}

__global__ void finalize_kernel(const double* sum, const unsigned* cnt, float* out) {
  out[0] = (*cnt > 0u) ? (float)(*sum / (double)(*cnt)) : 0.0f;
}

extern "C" void kernel_launch(void* const* d_in, const int* in_sizes, int n_in,
                              void* d_out, int out_size, void* d_ws, size_t ws_size,
                              hipStream_t stream) {
  const float* feat    = (const float*)d_in[0];
  const float* centers = (const float*)d_in[1];
  const int*   labels  = (const int*)d_in[2];

  // Workspace layout (all 16B-aligned); total ~13.8 MB (L2-resident on MI455X)
  char* ws = (char*)d_ws;
  double*   gsum = (double*)ws;                                   // 8 B
  unsigned* gcnt = (unsigned*)(ws + 8);                           // 4 B (+pad)
  float*    pos  = (float*)(ws + 16);                             // 32 KB
  size_t off = 16 + (size_t)B_ROWS * sizeof(float);               // 32784
  unsigned short* Fhi = (unsigned short*)(ws + off); off += (size_t)B_ROWS * DDIM * 2;
  unsigned short* Flo = (unsigned short*)(ws + off); off += (size_t)B_ROWS * DDIM * 2;
  unsigned short* Chi = (unsigned short*)(ws + off); off += (size_t)C_PAD * DDIM * 2;
  unsigned short* Clo = (unsigned short*)(ws + off);

  hipLaunchKernelGGL(init_accum, dim3(1), dim3(1), 0, stream, gsum, gcnt);
  hipLaunchKernelGGL(prep_kernel, dim3(B_ROWS), dim3(128), 0, stream,
                     feat, centers, labels, pos, Fhi, Flo);
  hipLaunchKernelGGL(cvt_centers_kernel, dim3(C_PAD * DDIM / 4 / 256), dim3(256), 0, stream,
                     centers, Chi, Clo);
  dim3 grid((C_COLS + BT - 1) / BT, B_ROWS / BT);
  hipLaunchKernelGGL(push_loss_kernel, grid, dim3(256), 0, stream,
                     Fhi, Flo, Chi, Clo, labels, pos, gsum, gcnt);
  hipLaunchKernelGGL(finalize_kernel, dim3(1), dim3(1), 0, stream,
                     gsum, gcnt, (float*)d_out);
}
